// Def_Block_S_12807592477450
// MI455X (gfx1250) — matrix operand order repaired and checked by reading, not compiled, not run on hardware
//
#include <hip/hip_runtime.h>
#include <hip/hip_bf16.h>

typedef __attribute__((ext_vector_type(16))) _Float16 v16h;
typedef __attribute__((ext_vector_type(8)))  float    v8f;
typedef __attribute__((ext_vector_type(2)))  _Float16 h2v;

namespace {
constexpr int B_ = 8, C_ = 128, H_ = 64, W_ = 64;
constexpr int KTAPS = 9;
constexpr int QCHUNKS = KTAPS * (C_ / 32);        // 36 K-chunks of 32
constexpr int MTILES  = B_ * H_ * (W_ / 16);      // 2048 tiles of 16 positions
constexpr float EPS_ = 1e-5f;
}

union U32H2 { unsigned u; _Float16 h[2]; };

// ---------------- pre-pass kernels ----------------

// NCHW fp32 -> NHWC f16 (idx runs over the NHWC layout)
__global__ void k_cvt_nhwc(const float* __restrict__ x, _Float16* __restrict__ xh) {
    int idx = blockIdx.x * blockDim.x + threadIdx.x;
    if (idx >= B_ * H_ * W_ * C_) return;
    int c  = idx & (C_ - 1);
    int p  = idx >> 7;                    // (b*H+h)*W + w
    int b  = p / (H_ * W_);
    int hw = p - b * H_ * W_;
    xh[idx] = (_Float16)x[(b * C_ + c) * (H_ * W_) + hw];
}

// fold conv-bias + BN into per-channel scale/shift
__global__ void k_prep_bn(const float* __restrict__ g, const float* __restrict__ be,
                          const float* __restrict__ m, const float* __restrict__ v,
                          const float* __restrict__ convb,
                          float* __restrict__ scale, float* __restrict__ bias) {
    int i = threadIdx.x;
    if (i < C_) {
        float s = g[i] * rsqrtf(v[i] + EPS_);
        scale[i] = s;
        bias[i]  = (convb[i] - m[i]) * s + be[i];
    }
}

// Pack OIHW fp32 weights into per-wave B-fragment order:
// wpack[((ntile*36+q)*32+lane)*8+v] = {f16 w[r][n], f16 w[r+1][n]}
// with r = tap*C + c, c = (q%4)*32 + (lane>=16?8:0) + (v>=4?16:0) + (v&3)*2, n = ntile*16 + lane%16
__global__ void k_pack_w(const float* __restrict__ wsrc, unsigned* __restrict__ wpack,
                         int Co, int ntiles) {
    int idx = blockIdx.x * blockDim.x + threadIdx.x;
    if (idx >= ntiles * QCHUNKS * 256) return;
    int v    = idx & 7;
    int lane = (idx >> 3) & 31;
    int q    = (idx >> 8) % QCHUNKS;
    int nt   = idx / (QCHUNKS * 256);
    int k  = q >> 2;
    int cc = q & 3;
    int n  = nt * 16 + (lane & 15);
    int c  = cc * 32 + ((lane >> 4) & 1) * 8 + ((v >> 2) * 16) + ((v & 3) * 2);
    U32H2 u;
    float w0f = 0.f, w1f = 0.f;
    if (n < Co) {
        w0f = wsrc[(n * C_ + c)     * KTAPS + k];
        w1f = wsrc[(n * C_ + c + 1) * KTAPS + k];
    }
    u.h[0] = (_Float16)w0f;
    u.h[1] = (_Float16)w1f;
    wpack[idx] = u.u;
}

// ---------------- fragment helpers ----------------

// A fragment (16x32 f16): lane L, VGPR v -> M=L%16, K=(v>=4?16:0)+(L>=16?8:0)+(v&3)*2
__device__ __forceinline__ v16h load_a_conv(const _Float16* __restrict__ act,
                                            int b, int y, int w0, int dx,
                                            int c0, int lane) {
    v16h a;
#pragma unroll
    for (int i = 0; i < 16; ++i) a[i] = (_Float16)0.f;
    if (y >= 0 && y < H_) {
        int m  = lane & 15;
        int xx = w0 + m + dx;
        if (xx >= 0 && xx < W_) {
            const _Float16* base =
                act + (((b * H_ + y) * W_ + xx) * C_) + c0 + ((lane >> 4) & 1) * 8;
#pragma unroll
            for (int v = 0; v < 8; ++v) {
                int coff = ((v >> 2) * 16) + ((v & 3) * 2);
                h2v p = *reinterpret_cast<const h2v*>(base + coff);
                a[2 * v]     = p[0];
                a[2 * v + 1] = p[1];
            }
        }
    }
    return a;
}

__device__ __forceinline__ v16h load_b_frag(const unsigned* __restrict__ wpack,
                                            int ntile, int q, int lane) {
    const uint4* p4 = reinterpret_cast<const uint4*>(
        wpack + (((ntile * QCHUNKS) + q) * 32 + lane) * 8);
    uint4 lo = p4[0], hi = p4[1];
    v16h bf; U32H2 t;
    t.u = lo.x; bf[0]  = t.h[0]; bf[1]  = t.h[1];
    t.u = lo.y; bf[2]  = t.h[0]; bf[3]  = t.h[1];
    t.u = lo.z; bf[4]  = t.h[0]; bf[5]  = t.h[1];
    t.u = lo.w; bf[6]  = t.h[0]; bf[7]  = t.h[1];
    t.u = hi.x; bf[8]  = t.h[0]; bf[9]  = t.h[1];
    t.u = hi.y; bf[10] = t.h[0]; bf[11] = t.h[1];
    t.u = hi.z; bf[12] = t.h[0]; bf[13] = t.h[1];
    t.u = hi.w; bf[14] = t.h[0]; bf[15] = t.h[1];
    return bf;
}

// ---------------- stage 1: conv1 + BN1 + ReLU -> y1 (NHWC f16) ----------------

__global__ void __launch_bounds__(256) k_conv1_wmma(
        const _Float16* __restrict__ xh, const unsigned* __restrict__ wpack,
        const float* __restrict__ scale, const float* __restrict__ bias,
        _Float16* __restrict__ y1) {
    int mtile = blockIdx.x;
    int ntile = threadIdx.x >> 5;
    int lane  = threadIdx.x & 31;
    int b  = mtile >> 8;
    int r  = mtile & 255;
    int h  = r >> 2;
    int w0 = (r & 3) * 16;

    v8f acc;
#pragma unroll
    for (int i = 0; i < 8; ++i) acc[i] = 0.f;

#pragma unroll 1
    for (int k = 0; k < KTAPS; ++k) {
        int dy = k / 3 - 1, dx = k % 3 - 1;
        int y = h + dy;
#pragma unroll
        for (int cc = 0; cc < 4; ++cc) {
            v16h a  = load_a_conv(xh, b, y, w0, dx, cc * 32, lane);
            v16h bf = load_b_frag(wpack, ntile, k * 4 + cc, lane);
            acc = __builtin_amdgcn_wmma_f32_16x16x32_f16(false, a, false, bf,
                                                         (short)0, acc, false, false);
        }
    }

    int n = ntile * 16 + (lane & 15);
    float s = scale[n], bi = bias[n];
    int mbase = ((lane >> 4) & 1) * 8;
#pragma unroll
    for (int t = 0; t < 8; ++t) {
        float vv = acc[t] * s + bi;
        vv = vv > 0.f ? vv : 0.f;
        y1[((b * H_ + h) * W_ + w0 + mbase + t) * C_ + n] = (_Float16)vv;
    }
}

// ---------------- stage 2: offset conv (Co = 18, fp32 out) ----------------

__global__ void __launch_bounds__(64) k_offconv_wmma(
        const _Float16* __restrict__ y1, const unsigned* __restrict__ wpack,
        const float* __restrict__ offb, float* __restrict__ off) {
    int mtile = blockIdx.x;
    int ntile = threadIdx.x >> 5;   // 0..1
    int lane  = threadIdx.x & 31;
    int b  = mtile >> 8;
    int r  = mtile & 255;
    int h  = r >> 2;
    int w0 = (r & 3) * 16;

    v8f acc;
#pragma unroll
    for (int i = 0; i < 8; ++i) acc[i] = 0.f;

#pragma unroll 1
    for (int k = 0; k < KTAPS; ++k) {
        int dy = k / 3 - 1, dx = k % 3 - 1;
        int y = h + dy;
#pragma unroll
        for (int cc = 0; cc < 4; ++cc) {
            v16h a  = load_a_conv(y1, b, y, w0, dx, cc * 32, lane);
            v16h bf = load_b_frag(wpack, ntile, k * 4 + cc, lane);
            acc = __builtin_amdgcn_wmma_f32_16x16x32_f16(false, a, false, bf,
                                                         (short)0, acc, false, false);
        }
    }

    int n = ntile * 16 + (lane & 15);
    if (n < 2 * KTAPS) {
        float bi = offb[n];
        int mbase = ((lane >> 4) & 1) * 8;
#pragma unroll
        for (int t = 0; t < 8; ++t)
            off[((b * H_ + h) * W_ + w0 + mbase + t) * (2 * KTAPS) + n] = acc[t] + bi;
    }
}

// ------- stage 3: deformable conv (bilinear gather) + BN2 + ReLU -> NCHW fp32 -------

__global__ void __launch_bounds__(256) k_dconv_wmma(
        const _Float16* __restrict__ y1, const float* __restrict__ off,
        const unsigned* __restrict__ wpack, const float* __restrict__ scale,
        const float* __restrict__ bias, float* __restrict__ out) {
    int mtile = blockIdx.x;
    int ntile = threadIdx.x >> 5;
    int lane  = threadIdx.x & 31;
    int b  = mtile >> 8;
    int r  = mtile & 255;
    int h  = r >> 2;
    int w0 = (r & 3) * 16;
    int m  = lane & 15;                       // A-row owned by this lane
    int posm = (b * H_ + h) * W_ + (w0 + m);
    int half8 = ((lane >> 4) & 1) * 8;

    v8f acc;
#pragma unroll
    for (int i = 0; i < 8; ++i) acc[i] = 0.f;

#pragma unroll 1
    for (int k = 0; k < KTAPS; ++k) {
        // sample coordinates for this lane's row (same on both lane halves)
        float offy = off[posm * (2 * KTAPS) + 2 * k];
        float offx = off[posm * (2 * KTAPS) + 2 * k + 1];
        float py = offy + (float)(k / 3) + (float)(h - 1);
        float px = offx + (float)(k % 3) + (float)(w0 + m - 1);
        float y0f = floorf(py), x0f = floorf(px);
        float wy = py - y0f, wx = px - x0f;
        int iy0 = (int)y0f, ix0 = (int)x0f;
        int iy1 = iy0 + 1,  ix1 = ix0 + 1;
        bool vy0 = (iy0 >= 0) && (iy0 < H_);
        bool vy1 = (iy1 >= 0) && (iy1 < H_);
        bool vx0 = (ix0 >= 0) && (ix0 < W_);
        bool vx1 = (ix1 >= 0) && (ix1 < W_);
        float c00 = (1.f - wy) * (1.f - wx) * ((vy0 && vx0) ? 1.f : 0.f);
        float c01 = (1.f - wy) * wx         * ((vy0 && vx1) ? 1.f : 0.f);
        float c10 = wy * (1.f - wx)         * ((vy1 && vx0) ? 1.f : 0.f);
        float c11 = wy * wx                 * ((vy1 && vx1) ? 1.f : 0.f);
        int cy0 = min(max(iy0, 0), H_ - 1);
        int cy1 = min(max(iy1, 0), H_ - 1);
        int cx0 = min(max(ix0, 0), W_ - 1);
        int cx1 = min(max(ix1, 0), W_ - 1);
        const _Float16* p00 = y1 + ((b * H_ + cy0) * W_ + cx0) * C_;
        const _Float16* p01 = y1 + ((b * H_ + cy0) * W_ + cx1) * C_;
        const _Float16* p10 = y1 + ((b * H_ + cy1) * W_ + cx0) * C_;
        const _Float16* p11 = y1 + ((b * H_ + cy1) * W_ + cx1) * C_;

#pragma unroll
        for (int cc = 0; cc < 4; ++cc) {
            v16h a;
#pragma unroll
            for (int v = 0; v < 8; ++v) {
                int c = cc * 32 + half8 + ((v >> 2) * 16) + ((v & 3) * 2);
                h2v q00 = *reinterpret_cast<const h2v*>(p00 + c);
                h2v q01 = *reinterpret_cast<const h2v*>(p01 + c);
                h2v q10 = *reinterpret_cast<const h2v*>(p10 + c);
                h2v q11 = *reinterpret_cast<const h2v*>(p11 + c);
                float lo = c00 * (float)q00[0] + c01 * (float)q01[0]
                         + c10 * (float)q10[0] + c11 * (float)q11[0];
                float hi = c00 * (float)q00[1] + c01 * (float)q01[1]
                         + c10 * (float)q10[1] + c11 * (float)q11[1];
                a[2 * v]     = (_Float16)lo;
                a[2 * v + 1] = (_Float16)hi;
            }
            v16h bf = load_b_frag(wpack, ntile, k * 4 + cc, lane);
            acc = __builtin_amdgcn_wmma_f32_16x16x32_f16(false, a, false, bf,
                                                         (short)0, acc, false, false);
        }
    }

    int o = ntile * 16 + (lane & 15);
    float s = scale[o], bi = bias[o];
    // lane's 8 accumulators are 8 consecutive W positions -> coalesced fp32 stores
    float* obase = out + ((b * C_ + o) * H_ + h) * W_ + w0 + half8;
#pragma unroll
    for (int t = 0; t < 8; ++t) {
        float vv = acc[t] * s + bi;
        obase[t] = vv > 0.f ? vv : 0.f;
    }
}

// ---------------- host launcher ----------------

extern "C" void kernel_launch(void* const* d_in, const int* in_sizes, int n_in,
                              void* d_out, int out_size, void* d_ws, size_t ws_size,
                              hipStream_t stream) {
    const float* x       = (const float*)d_in[0];
    const float* conv1_w = (const float*)d_in[1];
    const float* conv1_b = (const float*)d_in[2];
    const float* bn1_g   = (const float*)d_in[3];
    const float* bn1_b   = (const float*)d_in[4];
    const float* bn1_m   = (const float*)d_in[5];
    const float* bn1_v   = (const float*)d_in[6];
    const float* off_w   = (const float*)d_in[7];
    const float* off_b   = (const float*)d_in[8];
    const float* dconv_w = (const float*)d_in[9];
    const float* dconv_b = (const float*)d_in[10];
    const float* bn2_g   = (const float*)d_in[11];
    const float* bn2_b   = (const float*)d_in[12];
    const float* bn2_m   = (const float*)d_in[13];
    const float* bn2_v   = (const float*)d_in[14];

    char* ws = (char*)d_ws;
    auto carve = [&](size_t bytes) -> char* {
        char* p = ws;
        ws += (bytes + 255) & ~size_t(255);
        return p;
    };
    const size_t npix = (size_t)B_ * H_ * W_;
    _Float16* xh     = (_Float16*)carve(npix * C_ * sizeof(_Float16));   // x, NHWC f16
    _Float16* y1     = (_Float16*)carve(npix * C_ * sizeof(_Float16));   // stage1 out
    float*    offs   = (float*)   carve(npix * 2 * KTAPS * sizeof(float));
    unsigned* wp1    = (unsigned*)carve((size_t)8 * QCHUNKS * 256 * 4);
    unsigned* wpo    = (unsigned*)carve((size_t)2 * QCHUNKS * 256 * 4);
    unsigned* wp2    = (unsigned*)carve((size_t)8 * QCHUNKS * 256 * 4);
    float*    scale1 = (float*)carve(C_ * sizeof(float));
    float*    bias1  = (float*)carve(C_ * sizeof(float));
    float*    scale2 = (float*)carve(C_ * sizeof(float));
    float*    bias2  = (float*)carve(C_ * sizeof(float));
    (void)ws_size; (void)in_sizes; (void)n_in; (void)out_size;

    int tot = B_ * H_ * W_ * C_;
    k_cvt_nhwc<<<(tot + 255) / 256, 256, 0, stream>>>(x, xh);
    k_prep_bn<<<1, 128, 0, stream>>>(bn1_g, bn1_b, bn1_m, bn1_v, conv1_b, scale1, bias1);
    k_prep_bn<<<1, 128, 0, stream>>>(bn2_g, bn2_b, bn2_m, bn2_v, dconv_b, scale2, bias2);
    k_pack_w<<<(8 * QCHUNKS * 256 + 255) / 256, 256, 0, stream>>>(conv1_w, wp1, C_, 8);
    k_pack_w<<<(2 * QCHUNKS * 256 + 255) / 256, 256, 0, stream>>>(off_w, wpo, 2 * KTAPS, 2);
    k_pack_w<<<(8 * QCHUNKS * 256 + 255) / 256, 256, 0, stream>>>(dconv_w, wp2, C_, 8);

    k_conv1_wmma<<<MTILES, 256, 0, stream>>>(xh, wp1, scale1, bias1, y1);
    k_offconv_wmma<<<MTILES, 64, 0, stream>>>(y1, wpo, off_b, offs);
    k_dconv_wmma<<<MTILES, 256, 0, stream>>>(y1, offs, wp2, scale2, bias2, (float*)d_out);
}
